// Net_83674552861196
// MI455X (gfx1250) — matrix operand order repaired and checked by reading, not compiled, not run on hardware
//
#include <hip/hip_runtime.h>
#include <hip/hip_bf16.h>
#include <math.h>

// ---------------------------------------------------------------------------
// MoE net for MI455X (gfx1250, wave32, WMMA).
// Dense layers run as bf16 WMMA GEMMs (v_wmma_f32_16x16x32_bf16), each wave
// owning a 16x32 C strip (2 N-tiles, shared A fragment, 2 accumulators),
// f32 accumulation, compile-time-specialized epilogues. Convs are direct f32
// VALU with fused BN/ReLU/maxpool. Workspace use ~230 MB.
// ---------------------------------------------------------------------------

typedef __bf16 bf16_t;
typedef __attribute__((ext_vector_type(8)))  bf16_t bf16x8;
typedef __attribute__((ext_vector_type(16))) bf16_t bf16x16;
typedef __attribute__((ext_vector_type(8)))  float  v8f;

#define BATCH   4096
#define FDIM    3920
#define FPAD    3936        // 3920 padded to multiple of 32 (zero columns)
#define NEXP    8
#define HID     128
#define HID2    64
#define ODIM    1000
#define OPAD    1024        // 1000 padded to multiple of 32 (zero weight rows)
#define GH      64

// GEMM compile-time flags
#define F_RELU   1
#define F_OUTBF  2
#define F_GSCALE 4

// ---------------------------------------------------------------------------
// small init: zero BN stats accumulators + zero the padded feat columns
// ---------------------------------------------------------------------------
__global__ void zero_misc(float* stats, bf16_t* feat) {
    int i = blockIdx.x * 256 + threadIdx.x;
    if (i < 64) stats[i] = 0.f;
    int j = i - 64;
    if (j >= 0 && j < BATCH * 16) {
        int r = j >> 4, c = j & 15;
        feat[(size_t)r * FPAD + FDIM + c] = (bf16_t)0.f;
    }
}

// f32 -> bf16 with row/col zero padding (weights)
__global__ void convert_pad(bf16_t* dst, const float* src,
                            int rows_src, int cols_src,
                            int rows_dst, int cols_dst) {
    long i = (long)blockIdx.x * 256 + threadIdx.x;
    long total = (long)rows_dst * cols_dst;
    if (i >= total) return;
    int r = (int)(i / cols_dst), c = (int)(i % cols_dst);
    float v = (r < rows_src && c < cols_src) ? src[(size_t)r * cols_src + c] : 0.f;
    dst[i] = (bf16_t)v;
}

// w3cat[o][e*64+g] = ew3[e][o][g]  (o<1000; rows 1000..1023 zero)
__global__ void w3cat_kernel(bf16_t* dst, const float* ew3) {
    long i = (long)blockIdx.x * 256 + threadIdx.x;
    if (i >= (long)OPAD * 512) return;
    int o = (int)(i / 512), c = (int)(i % 512);
    int e = c >> 6, k = c & 63;
    float v = (o < ODIM) ? ew3[((size_t)e * ODIM + o) * 64 + k] : 0.f;
    dst[i] = (bf16_t)v;
}

// ---------------------------------------------------------------------------
// conv1 batch-stats pass: recompute conv1, accumulate per-channel sum/sumsq
// ---------------------------------------------------------------------------
__global__ __launch_bounds__(256)
void conv1_stats(const float* __restrict__ x, const float* __restrict__ w,
                 const float* __restrict__ bconv, float* stats) {
    __shared__ float wc[75];
    __shared__ float rs[256], rs2[256];
    const int c = blockIdx.y;
    if (threadIdx.x < 75) wc[threadIdx.x] = w[c * 75 + threadIdx.x];
    __syncthreads();
    const float bias = bconv[c];
    const long total = (long)BATCH * 62 * 62;
    float s = 0.f, s2 = 0.f;
    for (long idx = (long)blockIdx.x * 256 + threadIdx.x; idx < total;
         idx += (long)gridDim.x * 256) {
        int n = (int)(idx / 3844), rem = (int)(idx % 3844);
        int oy = rem / 62, ox = rem % 62;
        const float* xb = x + (size_t)n * 3 * 4096;
        float acc = bias;
        #pragma unroll
        for (int ci = 0; ci < 3; ++ci)
            #pragma unroll
            for (int ky = 0; ky < 5; ++ky) {
                int iy = oy + ky - 1;
                if ((unsigned)iy < 64u)
                    #pragma unroll
                    for (int kx = 0; kx < 5; ++kx) {
                        int ix = ox + kx - 1;
                        if ((unsigned)ix < 64u)
                            acc += xb[ci * 4096 + iy * 64 + ix] * wc[ci * 25 + ky * 5 + kx];
                    }
            }
        s += acc; s2 += acc * acc;
    }
    rs[threadIdx.x] = s; rs2[threadIdx.x] = s2;
    __syncthreads();
    for (int st = 128; st > 0; st >>= 1) {
        if (threadIdx.x < st) { rs[threadIdx.x] += rs[threadIdx.x + st];
                                rs2[threadIdx.x] += rs2[threadIdx.x + st]; }
        __syncthreads();
    }
    if (threadIdx.x == 0) {
        atomicAdd(&stats[c], rs[0]);
        atomicAdd(&stats[10 + c], rs2[0]);
    }
}

__global__ void bn_finalize(float* stats, const float* g, const float* b) {
    int c = threadIdx.x;
    if (c < 10) {
        const float cnt = 15745024.f;      // 4096*62*62
        float mean = stats[c] / cnt;
        float var  = stats[10 + c] / cnt - mean * mean;
        float sc = g[c] * rsqrtf(var + 1e-5f);
        stats[20 + c] = sc;
        stats[30 + c] = b[c] - mean * sc;
    }
}

// ---------------------------------------------------------------------------
// conv1 + BN + ReLU + 2x2 maxpool fused: one thread per pooled output
// ---------------------------------------------------------------------------
__global__ __launch_bounds__(256)
void conv1_pool(const float* __restrict__ x, const float* __restrict__ w,
                const float* __restrict__ bconv, const float* __restrict__ stats,
                float* __restrict__ pool1) {
    size_t idx = (size_t)blockIdx.x * 256 + threadIdx.x;
    if (idx >= (size_t)BATCH * 10 * 31 * 31) return;
    int p = (int)(idx % 961); int tmp = (int)(idx / 961);
    int c = tmp % 10, n = tmp / 10;
    int py = p / 31, px = p % 31;
    const float* xb = x + (size_t)n * 3 * 4096;
    const float* wc = w + c * 75;
    const float bias = bconv[c];
    const float sc = stats[20 + c], sh = stats[30 + c];
    float mx = 0.f;   // ReLU output >= 0
    #pragma unroll
    for (int dy = 0; dy < 2; ++dy)
        #pragma unroll
        for (int dx = 0; dx < 2; ++dx) {
            int oy = 2 * py + dy, ox = 2 * px + dx;
            float acc = bias;
            #pragma unroll
            for (int ci = 0; ci < 3; ++ci)
                #pragma unroll
                for (int ky = 0; ky < 5; ++ky) {
                    int iy = oy + ky - 1;
                    if ((unsigned)iy < 64u)
                        #pragma unroll
                        for (int kx = 0; kx < 5; ++kx) {
                            int ix = ox + kx - 1;
                            if ((unsigned)ix < 64u)
                                acc += xb[ci * 4096 + iy * 64 + ix] * wc[ci * 25 + ky * 5 + kx];
                        }
                }
            float r = acc * sc + sh;
            r = r > 0.f ? r : 0.f;
            mx = r > mx ? r : mx;
        }
    pool1[idx] = mx;
}

// conv2 + ReLU + 2x2 maxpool fused; writes bf16 features (padded row stride)
__global__ __launch_bounds__(256)
void conv2_pool(const float* __restrict__ pool1, const float* __restrict__ w,
                const float* __restrict__ bconv, bf16_t* __restrict__ feat) {
    size_t idx = (size_t)blockIdx.x * 256 + threadIdx.x;
    if (idx >= (size_t)BATCH * 20 * 14 * 14) return;
    int p = (int)(idx % 196); int tmp = (int)(idx / 196);
    int c = tmp % 20, n = tmp / 20;
    int py = p / 14, px = p % 14;
    const float* pb = pool1 + (size_t)n * 10 * 961;
    const float* wc = w + c * 250;
    const float bias = bconv[c];
    float mx = 0.f;
    #pragma unroll
    for (int dy = 0; dy < 2; ++dy)
        #pragma unroll
        for (int dx = 0; dx < 2; ++dx) {
            int oy = 2 * py + dy, ox = 2 * px + dx;   // <= 27 (< 29)
            float acc = bias;
            for (int ci = 0; ci < 10; ++ci)
                #pragma unroll
                for (int ky = 0; ky < 5; ++ky) {
                    int iy = oy + ky - 1;
                    if ((unsigned)iy < 31u)
                        #pragma unroll
                        for (int kx = 0; kx < 5; ++kx) {
                            int ix = ox + kx - 1;
                            if ((unsigned)ix < 31u)
                                acc += pb[ci * 961 + iy * 31 + ix] * wc[ci * 25 + ky * 5 + kx];
                        }
                }
            float r = acc > 0.f ? acc : 0.f;
            mx = r > mx ? r : mx;
        }
    feat[(size_t)n * FPAD + c * 196 + py * 14 + px] = (bf16_t)mx;
}

// ---------------------------------------------------------------------------
// bf16 WMMA GEMM: one 16x(16*NT) C strip per wave, NT accumulators sharing
// one A fragment. Compile-time FLAGS specialize the epilogue (no runtime
// branch code in the K loop).
//   A: [M x lda] bf16 (row-major, K contiguous)  (+ aZ per blockIdx.z)
//   B: [Npad x K] bf16 (row = output neuron, K contiguous) (+ bZ per z)
//   C: [M x ldc] f32 or bf16, tile strip written at column offset cZ*z
// Fragment layouts per CDNA5 ISA §7.12.2 (wave32):
//   A 16x32: lanes 0-15 = rows M, halves {0-7}->K 0-7, {8-15}->K 16-23;
//            lanes 16-31 same rows, K offset +8.
//   B 32x16: lane%16 = N column, same K mapping as A (halves {0-7}->K 0-7, {8-15}->K 16-23; lanes>=16: K +8).
//   C/D:     v[r] -> M = r + 8*(lane>=16), N = lane%16.
// ---------------------------------------------------------------------------
template<int FLAGS, int NT>
__global__ __launch_bounds__(256)
void gemm_bf16_wmma(const bf16_t* __restrict__ A, long long aZ, int lda,
                    const bf16_t* __restrict__ B, long long bZ, int K,
                    const float* __restrict__ bias, long long biasZ,
                    void* __restrict__ C, long long cZ, int ldc,
                    const float* __restrict__ gates,
                    int Nreal, int tilesNgrp) {
    const int z    = blockIdx.z;
    const int wave = threadIdx.x >> 5;
    const int lane = threadIdx.x & 31;
    const int gwid = blockIdx.x * 8 + wave;
    const int tM  = gwid / tilesNgrp;
    const int tNg = gwid % tilesNgrp;
    const int m16 = lane & 15;
    const int hi  = (lane >= 16) ? 1 : 0;

    const bf16_t* Arow = A + (size_t)aZ * z + (size_t)(tM * 16 + m16) * lda;
    const bf16_t* Brow = B + (size_t)bZ * z + (size_t)(tNg * NT * 16 + m16) * K;
    const int koffA = hi * 8;
    const int koffB = hi * 8;

    v8f acc[NT];
    #pragma unroll
    for (int t = 0; t < NT; ++t) acc[t] = (v8f){0.f,0.f,0.f,0.f,0.f,0.f,0.f,0.f};

    for (int kk = 0; kk < K; kk += 32) {
        union Frag { bf16x16 v; bf16x8 h[2]; } a, b[NT];
        a.h[0] = *(const bf16x8*)(Arow + kk + koffA);
        a.h[1] = *(const bf16x8*)(Arow + kk + koffA + 16);
        #pragma unroll
        for (int t = 0; t < NT; ++t) {
            const bf16_t* bp = Brow + (size_t)t * 16 * K + kk + koffB;
            b[t].h[0] = *(const bf16x8*)(bp);
            b[t].h[1] = *(const bf16x8*)(bp + 16);
        }
        #pragma unroll
        for (int t = 0; t < NT; ++t)
            acc[t] = __builtin_amdgcn_wmma_f32_16x16x32_bf16(
                         false, a.v, false, b[t].v, (short)0, acc[t], false, false);
    }

    const int mBase = tM * 16 + hi * 8;
    #pragma unroll
    for (int t = 0; t < NT; ++t) {
        const int nCol = (tNg * NT + t) * 16 + m16;
        const bool nOk = nCol < Nreal;
        float bv = 0.f;
        if (bias != nullptr && nOk) bv = bias[(size_t)biasZ * z + nCol];
        #pragma unroll
        for (int r = 0; r < 8; ++r) {
            const int mRow = mBase + r;
            float v = acc[t][r] + bv;
            if constexpr (FLAGS & F_RELU)   v = v > 0.f ? v : 0.f;
            if constexpr (FLAGS & F_GSCALE) v *= gates[(size_t)mRow * NEXP + z];
            if (nOk) {
                size_t ci = (size_t)mRow * ldc + (size_t)cZ * z + nCol;
                if constexpr (FLAGS & F_OUTBF) ((bf16_t*)C)[ci] = (bf16_t)v;
                else                           ((float*)C)[ci]  = v;
            }
        }
    }
}

// ---------------------------------------------------------------------------
// Top-2 gate: per row, pick top-2 logits (ties -> lower index), softmax(2)
// ---------------------------------------------------------------------------
__global__ void gate_topk(const float* __restrict__ logits, float* __restrict__ gates) {
    int b = blockIdx.x * 256 + threadIdx.x;
    if (b >= BATCH) return;
    float v[NEXP];
    #pragma unroll
    for (int e = 0; e < NEXP; ++e) v[e] = logits[(size_t)b * 16 + e];
    int i1 = 0;
    #pragma unroll
    for (int e = 1; e < NEXP; ++e) if (v[e] > v[i1]) i1 = e;
    int i2 = (i1 == 0) ? 1 : 0;
    #pragma unroll
    for (int e = 0; e < NEXP; ++e) if (e != i1 && v[e] > v[i2]) i2 = e;
    float e2 = expf(v[i2] - v[i1]);
    float g1 = 1.f / (1.f + e2);
    float g2 = e2 * g1;
    #pragma unroll
    for (int e = 0; e < NEXP; ++e) gates[(size_t)b * NEXP + e] = 0.f;
    gates[(size_t)b * NEXP + i1] = g1;
    gates[(size_t)b * NEXP + i2] = g2;
}

// ---------------------------------------------------------------------------
// Final: out += sum_e gates*eb3 ; t=|out| ; log_softmax over 1000 classes
// ---------------------------------------------------------------------------
__global__ __launch_bounds__(256)
void final_logsoftmax(const float* __restrict__ outb, const float* __restrict__ gates,
                      const float* __restrict__ eb3, float* __restrict__ out) {
    const int b = blockIdx.x, tid = threadIdx.x;
    __shared__ float red[256];
    float g[NEXP];
    #pragma unroll
    for (int e = 0; e < NEXP; ++e) g[e] = gates[(size_t)b * NEXP + e];
    float tv[4];
    float mx = -1e30f;
    #pragma unroll
    for (int i = 0; i < 4; ++i) {
        int o = tid + i * 256;
        if (o < ODIM) {
            float v = outb[(size_t)b * OPAD + o];
            float bb = 0.f;
            #pragma unroll
            for (int e = 0; e < NEXP; ++e) bb += g[e] * eb3[(size_t)e * ODIM + o];
            v = fabsf(v + bb);
            tv[i] = v;
            mx = v > mx ? v : mx;
        } else tv[i] = -1e30f;
    }
    red[tid] = mx; __syncthreads();
    for (int s = 128; s > 0; s >>= 1) {
        if (tid < s) { float o = red[tid + s]; if (o > red[tid]) red[tid] = o; }
        __syncthreads();
    }
    mx = red[0]; __syncthreads();
    float sum = 0.f;
    #pragma unroll
    for (int i = 0; i < 4; ++i)
        if (tid + i * 256 < ODIM) sum += expf(tv[i] - mx);
    red[tid] = sum; __syncthreads();
    for (int s = 128; s > 0; s >>= 1) {
        if (tid < s) red[tid] += red[tid + s];
        __syncthreads();
    }
    float lse = logf(red[0]);
    #pragma unroll
    for (int i = 0; i < 4; ++i) {
        int o = tid + i * 256;
        if (o < ODIM) out[(size_t)b * ODIM + o] = tv[i] - mx - lse;
    }
}

// ---------------------------------------------------------------------------
extern "C" void kernel_launch(void* const* d_in, const int* in_sizes, int n_in,
                              void* d_out, int out_size, void* d_ws, size_t ws_size,
                              hipStream_t stream) {
    const float* x       = (const float*)d_in[0];
    const float* conv1_w = (const float*)d_in[1];
    const float* conv1_b = (const float*)d_in[2];
    const float* bn_g    = (const float*)d_in[3];
    const float* bn_b    = (const float*)d_in[4];
    const float* conv2_w = (const float*)d_in[5];
    const float* conv2_b = (const float*)d_in[6];
    const float* gate_w1 = (const float*)d_in[7];
    const float* gate_b1 = (const float*)d_in[8];
    const float* gate_w2 = (const float*)d_in[9];
    const float* gate_b2 = (const float*)d_in[10];
    const float* ew1     = (const float*)d_in[11];
    const float* eb1     = (const float*)d_in[12];
    const float* ew2     = (const float*)d_in[13];
    const float* eb2     = (const float*)d_in[14];
    const float* ew3     = (const float*)d_in[15];
    const float* eb3     = (const float*)d_in[16];
    float* out           = (float*)d_out;

    // workspace carve-up (all offsets 32B-aligned for bf16x16 loads)
    char* ws = (char*)d_ws;
    size_t off = 0;
    float* stats  = (float*)(ws + off);                off += 256;
    float* pool1  = (float*)(ws + off);                off += (size_t)BATCH*10*31*31*4;  // 157.45 MB
    bf16_t* feat  = (bf16_t*)(ws + off);               off += (size_t)BATCH*FPAD*2;      // 32.24 MB
    bf16_t* gw1b  = (bf16_t*)(ws + off);               off += (size_t)GH*FPAD*2;
    bf16_t* gw2b  = (bf16_t*)(ws + off);               off += (size_t)16*GH*2;
    bf16_t* ew1b  = (bf16_t*)(ws + off);               off += (size_t)NEXP*HID*FPAD*2;   // 8.06 MB
    bf16_t* ew2b  = (bf16_t*)(ws + off);               off += (size_t)NEXP*HID2*HID*2;
    bf16_t* w3cat = (bf16_t*)(ws + off);               off += (size_t)OPAD*512*2;
    bf16_t* gact  = (bf16_t*)(ws + off);               off += (size_t)BATCH*GH*2;
    float* logits = (float*)(ws + off);                off += (size_t)BATCH*16*4;
    float* gates  = (float*)(ws + off);                off += (size_t)BATCH*NEXP*4;
    bf16_t* h1    = (bf16_t*)(ws + off);               off += (size_t)BATCH*NEXP*HID*2;  // 8.39 MB
    bf16_t* h2    = (bf16_t*)(ws + off);               off += (size_t)BATCH*NEXP*HID2*2;
    float* outb   = (float*)(ws + off);                off += (size_t)BATCH*OPAD*4;      // 16.78 MB

    // init + weight conversion (every call: deterministic, no cached state)
    zero_misc<<<257, 256, 0, stream>>>(stats, feat);
    convert_pad<<<(GH*FPAD + 255)/256, 256, 0, stream>>>(gw1b, gate_w1, GH, FDIM, GH, FPAD);
    convert_pad<<<(16*GH + 255)/256, 256, 0, stream>>>(gw2b, gate_w2, NEXP, GH, 16, GH);
    convert_pad<<<(NEXP*HID*FPAD + 255)/256, 256, 0, stream>>>(ew1b, ew1, NEXP*HID, FDIM, NEXP*HID, FPAD);
    convert_pad<<<(NEXP*HID2*HID + 255)/256, 256, 0, stream>>>(ew2b, ew2, NEXP*HID2, HID, NEXP*HID2, HID);
    w3cat_kernel<<<(OPAD*512 + 255)/256, 256, 0, stream>>>(w3cat, ew3);

    // feature extractor
    conv1_stats<<<dim3(256, 10), 256, 0, stream>>>(x, conv1_w, conv1_b, stats);
    bn_finalize<<<1, 32, 0, stream>>>(stats, bn_g, bn_b);
    conv1_pool<<<(BATCH*10*31*31)/256, 256, 0, stream>>>(x, conv1_w, conv1_b, stats, pool1);
    conv2_pool<<<(BATCH*20*14*14)/256, 256, 0, stream>>>(pool1, conv2_w, conv2_b, feat);

    // gate: relu(feat@W1^T+b1) (feat>=0 so |feat|==feat) -> logits -> top-2
    gemm_bf16_wmma<F_RELU|F_OUTBF, 2><<<dim3(64,1,1), 256, 0, stream>>>(
        feat, 0, FPAD, gw1b, 0, FPAD, gate_b1, 0,
        gact, 0, GH, nullptr, GH, 2);
    gemm_bf16_wmma<0, 1><<<dim3(32,1,1), 256, 0, stream>>>(
        gact, 0, GH, gw2b, 0, GH, gate_b2, 0,
        logits, 0, 16, nullptr, NEXP, 1);
    gate_topk<<<BATCH/256, 256, 0, stream>>>(logits, gates);

    // experts: fc1 (per-expert), fc2 (per-expert, gate-scaled), fc3 (fused)
    gemm_bf16_wmma<F_RELU|F_OUTBF, 2><<<dim3(128,1,NEXP), 256, 0, stream>>>(
        feat, 0, FPAD, ew1b, (long long)HID*FPAD, FPAD, eb1, HID,
        h1, HID, NEXP*HID, nullptr, HID, 4);
    gemm_bf16_wmma<F_RELU|F_OUTBF|F_GSCALE, 2><<<dim3(64,1,NEXP), 256, 0, stream>>>(
        h1, HID, NEXP*HID, ew2b, (long long)HID2*HID, HID, eb2, HID2,
        h2, HID2, NEXP*HID2, gates, HID2, 2);
    gemm_bf16_wmma<0, 2><<<dim3(1024,1,1), 256, 0, stream>>>(
        h2, 0, NEXP*HID2, w3cat, 0, 512, nullptr, 0,
        outb, 0, OPAD, nullptr, ODIM, OPAD/32);

    // combine gated bias, |.|, log-softmax
    final_logsoftmax<<<BATCH, 256, 0, stream>>>(outb, gates, eb3, out);
}
